// SelfMultiheadAttention_6055903887449
// MI455X (gfx1250) — matrix operand order repaired and checked by reading, not compiled, not run on hardware
//
#include <hip/hip_runtime.h>
#include <hip/hip_bf16.h>
#include <math.h>

// Problem constants (match reference)
#define B_ 2
#define S_ 4096
#define F_ 768
#define H_ 12
#define D_ 64

typedef __attribute__((ext_vector_type(16))) __bf16 bf16x16;
typedef __attribute__((ext_vector_type(8)))  __bf16 bf16x8;
typedef __attribute__((ext_vector_type(8)))  float  f32x8;
typedef __attribute__((ext_vector_type(4)))  unsigned int u32x4;
typedef __attribute__((ext_vector_type(8)))  int          i32x8;
typedef __attribute__((ext_vector_type(4)))  int          i32x4;

#if __has_builtin(__builtin_amdgcn_tensor_load_to_lds)
#define HAVE_TDM 1
#else
#define HAVE_TDM 0
#endif

__device__ __forceinline__ __bf16 cvt_bf(float v)  { return (__bf16)v; }
__device__ __forceinline__ __bf16 cvt_bf(__bf16 v) { return v; }

// ---------------------------------------------------------------------------
// GEMM: C[M,N] = A[M,K] @ W[K,N] + bias[N]
// A is fp32 or bf16 (converted to bf16 during LDS staging), W/bias fp32.
// Block: 256 threads (8 waves), tile 128x128, BK=32.
// Wave grid 2x4; wave tile 64x32 = 4x2 WMMA accumulators (8 wmma / k-step).
// ---------------------------------------------------------------------------
template<typename AT, bool OUT_BF16>
__global__ __launch_bounds__(256, 2)
void gemm_wmma(const AT* __restrict__ A, const float* __restrict__ W,
               const float* __restrict__ bias, void* __restrict__ Cout,
               int M, int N, int K)
{
  constexpr int BM = 128, BN = 128, BK = 32;
  __shared__ __align__(16) __bf16 As[BM][BK + 8];   // 128 x 40 halves
  __shared__ __align__(16) __bf16 Bs[BN][BK + 8];   // transposed: [n][k]

  const int t    = threadIdx.x;
  const int lane = t & 31;
  const int wid  = t >> 5;          // 0..7
  const int r    = lane & 15;       // row/col within 16
  const int h    = lane >> 4;       // lane half

  const int m0 = blockIdx.y * BM;
  const int n0 = blockIdx.x * BN;
  const int mw = (wid >> 2) * 64;   // wave row offset (2 rows of waves)
  const int nw = (wid & 3) * 32;    // wave col offset (4 cols of waves)

  f32x8 acc[4][2] = {};

  for (int k0 = 0; k0 < K; k0 += BK) {
    // Prefetch next tiles into cache while this one is staged/consumed.
    if (k0 + BK < K) {
      __builtin_prefetch(&A[(size_t)(m0 + (t >> 1)) * K + (k0 + BK) + (t & 1) * 16], 0, 0);
      __builtin_prefetch(&W[(size_t)(k0 + BK + (t >> 3)) * N + n0 + (t & 7) * 16], 0, 0);
    }
    __syncthreads();
    // Stage A tile 128x32 (coalesced; convert to bf16 on the fly)
    #pragma unroll
    for (int i = 0; i < 16; ++i) {
      int idx = i * 256 + t;               // 0..4095
      int row = idx >> 5, col = idx & 31;
      As[row][col] = cvt_bf(A[(size_t)(m0 + row) * K + (k0 + col)]);
    }
    // Stage B tile transposed: Bs[n][k] = W[k0+k][n0+n] (coalesced global reads)
    #pragma unroll
    for (int i = 0; i < 16; ++i) {
      int idx = i * 256 + t;               // 0..4095
      int k = idx >> 7, n = idx & 127;
      Bs[n][k] = cvt_bf(W[(size_t)(k0 + k) * N + (n0 + n)]);
    }
    __syncthreads();

    // A fragments: 16x32, lane r = row, half h selects K sub-block.
    bf16x16 af[4], bfg[2];
    #pragma unroll
    for (int tm = 0; tm < 4; ++tm) {
      const __bf16* ap = &As[mw + tm * 16 + r][0];
      bf16x8 lo = *(const bf16x8*)(ap + h * 8);        // K = h*8 .. h*8+7
      bf16x8 hi = *(const bf16x8*)(ap + 16 + h * 8);   // K = 16+h*8 ..
      #pragma unroll
      for (int j = 0; j < 8; ++j) { af[tm][j] = lo[j]; af[tm][8 + j] = hi[j]; }
    }
    // B fragments: 32x16, lane r = col N, half h selects K sub-block (same order as A).
    #pragma unroll
    for (int tn = 0; tn < 2; ++tn) {
      const __bf16* bp = &Bs[nw + tn * 16 + r][0];
      bf16x8 b0 = *(const bf16x8*)(bp + h * 8);
      bf16x8 b1 = *(const bf16x8*)(bp + 16 + h * 8);
      #pragma unroll
      for (int j = 0; j < 8; ++j) { bfg[tn][j] = b0[j]; bfg[tn][8 + j] = b1[j]; }
    }

    #pragma unroll
    for (int tm = 0; tm < 4; ++tm)
      #pragma unroll
      for (int tn = 0; tn < 2; ++tn)
        acc[tm][tn] = __builtin_amdgcn_wmma_f32_16x16x32_bf16(
            false, af[tm], false, bfg[tn], (short)0, acc[tm][tn], false, false);
  }

  // Epilogue: C/D layout element (M = v + 8h, N = r); add bias, store.
  #pragma unroll
  for (int tm = 0; tm < 4; ++tm)
    #pragma unroll
    for (int tn = 0; tn < 2; ++tn)
      #pragma unroll
      for (int v = 0; v < 8; ++v) {
        int row = m0 + mw + tm * 16 + v + 8 * h;
        int col = n0 + nw + tn * 16 + r;
        float val = acc[tm][tn][v] + bias[col];
        if constexpr (OUT_BF16)
          ((__bf16*)Cout)[(size_t)row * N + col] = (__bf16)val;
        else
          ((float*)Cout)[(size_t)row * N + col] = val;
      }
}

// ---------------------------------------------------------------------------
// Flash attention (causal + key padding mask), bf16 WMMA, fp32 softmax.
// Block: 128 threads (4 waves) = 64 query rows of one (b, head).
// Each wave owns a 16-row q tile; K/V streamed in 32-key chunks via LDS.
// K chunk staged by the Tensor Data Mover (2D tile DMA + LDS row padding).
// ---------------------------------------------------------------------------
__global__ __launch_bounds__(128, 2)
void attn_wmma(const __bf16* __restrict__ qkv,        // [B*S, 3F] bf16
               const unsigned char* __restrict__ pad, // [B, S] bool
               __bf16* __restrict__ out)              // [B*S, F] bf16
{
  constexpr int LDK = D_ + 8;   // 72 halves = 144B rows (TDM pad: 4 DWORDs / row)
  constexpr int LDP = 32 + 8;   // 40 halves
  __shared__ __align__(16) __bf16 Ks[32][LDK];        // K chunk, key-major
  __shared__ __align__(16) __bf16 Vt[D_][LDP];        // V chunk, dim-major (transposed)
  __shared__ __align__(16) __bf16 Ps[4][16][LDP];     // per-wave P transpose scratch

  const int t    = threadIdx.x;
  const int lane = t & 31;
  const int wv   = t >> 5;                  // wave id 0..3
  const int r    = lane & 15;
  const int h    = lane >> 4;

  const int qblk = blockIdx.x;              // 0..S/64-1
  const int bh   = blockIdx.y;              // 0..B*H-1
  const int b    = bh / H_, head = bh % H_;
  const int qb   = qblk * 64 + wv * 16;     // this wave's first query row

  const int rowstride = 3 * F_;             // 2304
  const __bf16* Qbase = qkv + (size_t)(b * S_) * rowstride + head * D_;
  const __bf16* Kbase = Qbase + F_;
  const __bf16* Vbase = Qbase + 2 * F_;

  // Preload Q as two A-fragments (dims 0..31 and 32..63)
  bf16x16 qf[2];
  #pragma unroll
  for (int g = 0; g < 2; ++g) {
    const __bf16* qp = Qbase + (size_t)(qb + r) * rowstride + g * 32;
    bf16x8 lo = *(const bf16x8*)(qp + h * 8);
    bf16x8 hi = *(const bf16x8*)(qp + 16 + h * 8);
    #pragma unroll
    for (int j = 0; j < 8; ++j) { qf[g][j] = lo[j]; qf[g][8 + j] = hi[j]; }
  }

  f32x8 o[4] = {};                          // O accumulator: 16 x 64
  float mrow[8], lrow[8];
  #pragma unroll
  for (int v = 0; v < 8; ++v) { mrow[v] = -INFINITY; lrow[v] = 0.f; }

  const int nchunk = 2 * qblk + 2;          // causal extent for this block
  for (int c = 0; c < nchunk; ++c) {
    const int kbase = c * 32;
    __syncthreads();

#if HAVE_TDM
    // --- K chunk via Tensor Data Mover: 2D tile 32x64 bf16, global row stride
    // 2304 elems, LDS destination padded 4 DWORDs per 32-DWORD row (-> LDK=72).
    if (wv == 0) {
      unsigned lds_off = (unsigned)(size_t)(const void*)&Ks[0][0];
      unsigned long long ga =
          (unsigned long long)(size_t)(const void*)(Kbase + (size_t)kbase * rowstride);
      u32x4 g0;
      g0[0] = 1u;                                   // count=1 (valid user D#)
      g0[1] = lds_off;                              // lds_addr
      g0[2] = (unsigned)ga;                         // global_addr[31:0]
      g0[3] = (unsigned)((ga >> 32) & 0x1FFFFFFull) // global_addr[56:32]
              | (2u << 30);                         // type=2 ("image")
      i32x8 g1;
      g1[0] = (int)((1u << 16)      // data_size = 1 (2 bytes)
              |     (1u << 20)      // pad_enable
              |     (4u << 22)      // pad_interval: 32 DWORDs (one 128B row)
              |     (3u << 25));    // pad_amount: 4 DWORDs (16B)
      g1[1] = (int)(64u << 16);     // tensor_dim0[15:0] = 64 (bits 63:48)
      g1[2] = (int)(32u << 16);     // tensor_dim1[15:0] = 32 (bits 95:80)
      g1[3] = (int)(64u << 16);     // tile_dim0 = 64     (bits 127:112)
      g1[4] = 32;                   // tile_dim1 = 32     (bits 143:128)
      g1[5] = 2304;                 // tensor_dim0_stride[31:0]
      g1[6] = 0;
      g1[7] = 0;
      i32x4 gz  = {0, 0, 0, 0};
      i32x8 gz8 = {0, 0, 0, 0, 0, 0, 0, 0};
      // 6-arg form (clang-23 / therock-10.0 toolchain)
      __builtin_amdgcn_tensor_load_to_lds(g0, g1, gz, gz, gz8, 0);
    }
#else
    // Manual K staging fallback (8-half vector copies, coalesced)
    #pragma unroll
    for (int i = 0; i < 2; ++i) {
      int idx = i * 128 + t;
      int key = idx >> 3, cc = idx & 7;
      bf16x8 kv = *(const bf16x8*)(Kbase + (size_t)(kbase + key) * rowstride + cc * 8);
      *(bf16x8*)&Ks[key][cc * 8] = kv;
    }
#endif
    // Stage V chunk transposed: Vt[dim][key] (TDM cannot transpose)
    #pragma unroll
    for (int i = 0; i < 2; ++i) {
      int idx = i * 128 + t;
      int key = idx >> 3, cc = idx & 7;
      bf16x8 vv = *(const bf16x8*)(Vbase + (size_t)(kbase + key) * rowstride + cc * 8);
      #pragma unroll
      for (int j = 0; j < 8; ++j) Vt[cc * 8 + j][key] = vv[j];
    }
#if HAVE_TDM
    if (wv == 0) __builtin_amdgcn_s_wait_tensorcnt(0);
#endif
    __syncthreads();

    if (kbase > qb + 15) continue;          // wave fully above diagonal (uniform)

    // Scores: Q(16x64) @ K^T(64x32) -> two 16x16 tiles
    f32x8 s[2] = {};
    #pragma unroll
    for (int kt = 0; kt < 2; ++kt)
      #pragma unroll
      for (int g = 0; g < 2; ++g) {
        const __bf16* kp = &Ks[kt * 16 + r][g * 32 + h * 8];
        bf16x8 b0 = *(const bf16x8*)(kp);
        bf16x8 b1 = *(const bf16x8*)(kp + 16);
        bf16x16 kfrag;
        #pragma unroll
        for (int j = 0; j < 8; ++j) { kfrag[j] = b0[j]; kfrag[8 + j] = b1[j]; }
        s[kt] = __builtin_amdgcn_wmma_f32_16x16x32_bf16(
            false, qf[g], false, kfrag, (short)0, s[kt], false, false);
      }

    // Scale 1/sqrt(D) + causal + padding mask
    #pragma unroll
    for (int kt = 0; kt < 2; ++kt) {
      int key_g = kbase + kt * 16 + r;
      bool padk = pad[b * S_ + key_g] != 0;
      #pragma unroll
      for (int v = 0; v < 8; ++v) {
        int q_g = qb + v + 8 * h;
        float val = s[kt][v] * 0.125f;
        if (key_g > q_g || padk) val = -INFINITY;
        s[kt][v] = val;
      }
    }

    // Online softmax: per-row (v) reductions across the 16 lanes of each half
    float corr[8];
    #pragma unroll
    for (int v = 0; v < 8; ++v) {
      float mx = fmaxf(s[0][v], s[1][v]);
      mx = fmaxf(mx, __shfl_xor(mx, 1));
      mx = fmaxf(mx, __shfl_xor(mx, 2));
      mx = fmaxf(mx, __shfl_xor(mx, 4));
      mx = fmaxf(mx, __shfl_xor(mx, 8));
      float mn = fmaxf(mrow[v], mx);
      float cf = __expf(mrow[v] - mn);
      float p0 = __expf(s[0][v] - mn);
      float p1 = __expf(s[1][v] - mn);
      s[0][v] = p0; s[1][v] = p1;
      float rs = p0 + p1;
      rs += __shfl_xor(rs, 1);
      rs += __shfl_xor(rs, 2);
      rs += __shfl_xor(rs, 4);
      rs += __shfl_xor(rs, 8);
      lrow[v] = lrow[v] * cf + rs;
      mrow[v] = mn;
      corr[v] = cf;
    }
    #pragma unroll
    for (int nb = 0; nb < 4; ++nb)
      #pragma unroll
      for (int v = 0; v < 8; ++v) o[nb][v] *= corr[v];

    // Transpose P (C/D layout -> A-fragment layout) through per-wave LDS scratch
    #pragma unroll
    for (int kt = 0; kt < 2; ++kt)
      #pragma unroll
      for (int v = 0; v < 8; ++v)
        Ps[wv][v + 8 * h][kt * 16 + r] = (__bf16)s[kt][v];

    asm volatile("s_wait_dscnt 0" ::: "memory");  // wave-private scratch RAW

    bf16x16 pf;
    {
      const __bf16* pp = &Ps[wv][r][0];
      bf16x8 lo = *(const bf16x8*)(pp + h * 8);
      bf16x8 hi = *(const bf16x8*)(pp + 16 + h * 8);
      #pragma unroll
      for (int j = 0; j < 8; ++j) { pf[j] = lo[j]; pf[8 + j] = hi[j]; }
    }

    // O += P(16x32) @ V(32x64): 4 WMMAs over dim tiles
    #pragma unroll
    for (int nb = 0; nb < 4; ++nb) {
      const __bf16* vp = &Vt[nb * 16 + r][h * 8];
      bf16x8 b0 = *(const bf16x8*)(vp);
      bf16x8 b1 = *(const bf16x8*)(vp + 16);
      bf16x16 vfrag;
      #pragma unroll
      for (int j = 0; j < 8; ++j) { vfrag[j] = b0[j]; vfrag[8 + j] = b1[j]; }
      o[nb] = __builtin_amdgcn_wmma_f32_16x16x32_bf16(
          false, pf, false, vfrag, (short)0, o[nb], false, false);
    }
  }

  // Normalize and store attention output (bf16) for the final projection
  #pragma unroll
  for (int v = 0; v < 8; ++v) {
    float inv = 1.0f / lrow[v];
    int row = b * S_ + qb + v + 8 * h;
    #pragma unroll
    for (int nb = 0; nb < 4; ++nb) {
      int col = head * D_ + nb * 16 + r;
      out[(size_t)row * F_ + col] = (__bf16)(o[nb][v] * inv);
    }
  }
}

// ---------------------------------------------------------------------------
extern "C" void kernel_launch(void* const* d_in, const int* in_sizes, int n_in,
                              void* d_out, int out_size, void* d_ws, size_t ws_size,
                              hipStream_t stream) {
  const float* x             = (const float*)d_in[0];
  const unsigned char* pmask = (const unsigned char*)d_in[1];
  const float* Wqkv          = (const float*)d_in[2];
  const float* bqkv          = (const float*)d_in[3];
  const float* Wout          = (const float*)d_in[4];
  const float* bout          = (const float*)d_in[5];
  float* out                 = (float*)d_out;

  const int M = B_ * S_;        // 8192
  __bf16* qkvb  = (__bf16*)d_ws;                                   // [M, 3F] bf16
  __bf16* attnb = (__bf16*)((char*)d_ws + (size_t)M * 3 * F_ * 2); // [M, F] bf16

  // 1) QKV projection (bf16 out, bias fused)
  dim3 g1((3 * F_) / 128, M / 128);
  gemm_wmma<float, true><<<g1, 256, 0, stream>>>(x, Wqkv, bqkv, (void*)qkvb,
                                                 M, 3 * F_, F_);
  // 2) Flash attention
  dim3 g2(S_ / 64, B_ * H_);
  attn_wmma<<<g2, 128, 0, stream>>>(qkvb, pmask, attnb);

  // 3) Output projection (fp32 out, bias fused)
  dim3 g3(F_ / 128, M / 128);
  gemm_wmma<__bf16, false><<<g3, 256, 0, stream>>>(attnb, Wout, bout, (void*)out,
                                                   M, F_, F_);
}
